// Classifier_56891136803038
// MI455X (gfx1250) — matrix operand order repaired and checked by reading, not compiled, not run on hardware
//
#include <hip/hip_runtime.h>
#include <hip/hip_bf16.h>

typedef __attribute__((ext_vector_type(16))) __bf16 v16bf;
typedef __attribute__((ext_vector_type(8)))  float  v8f;

static constexpr int BATCH = 64;
static constexpr int LR    = 729;   // real tokens (27*27)
static constexpr int LP    = 736;   // padded to 46*16
static constexpr int NT    = 46;    // L tiles of 16
static constexpr int D     = 256;   // dm = dk = dv
static constexpr int IMG   = 224;
static constexpr int GP    = 27;    // patches per image dim

#define LANE (threadIdx.x & 31)
#define WAVE (threadIdx.x >> 5)

// ---------------- WMMA helpers (CDNA5 16x16x32 BF16, f32 accum) -------------

__device__ __forceinline__ v8f wmma_bf16(v16bf a, v16bf b, v8f c) {
  // emits v_wmma_f32_16x16x32_bf16
  return __builtin_amdgcn_wmma_f32_16x16x32_bf16(false, a, false, b, (short)0, c,
                                                 false, false);
}

// A fragment (16x32, row-major source, stride lda). Layout used here:
// lanes 0-15 hold M=lane; VGPR r: K=2r..2r+1 (+16 for lanes 16-31),
// i.e. contiguous K order matching load_bT_frag.
__device__ __forceinline__ v16bf load_a_frag(const __bf16* A, int row0, int lda,
                                             int kof, int lane) {
  int m = (lane & 15) + row0, hh = lane >> 4;
  const __bf16* p = A + (size_t)m * lda + kof;
  v16bf a;
#pragma unroll
  for (int r = 0; r < 8; ++r) {
    int kb = 2 * r + 16 * hh;
    a[2 * r]     = p[kb];
    a[2 * r + 1] = p[kb + 1];
  }
  return a;
}

// Same A-fragment but gathering from an fp32 source (convert on load).
__device__ __forceinline__ v16bf load_a_frag_f32(const float* A, int row0, int lda,
                                                 int kof, int lane) {
  int m = (lane & 15) + row0, hh = lane >> 4;
  const float* p = A + (size_t)m * lda + kof;
  v16bf a;
#pragma unroll
  for (int r = 0; r < 8; ++r) {
    int kb = 2 * r + 16 * hh;
    a[2 * r]     = (__bf16)p[kb];
    a[2 * r + 1] = (__bf16)p[kb + 1];
  }
  return a;
}

// B fragment (32x16) where B[k][n] = W[n][k] (W row-major, stride ldw):
// lane n = lane&15; lanes 0-15 K=0..15, lanes 16-31 K=16..31 -> 16 contiguous
// elements of row n => one 32B vector load (two global_load_b128).
// Requires 32B alignment: ldw*2 must be a multiple of 32 (true for 256 and 736).
__device__ __forceinline__ v16bf load_bT_frag(const __bf16* W, int n0, int ldw,
                                              int kof, int lane) {
  int n = (lane & 15) + n0, hh = lane >> 4;
  return *(const v16bf*)(W + (size_t)n * ldw + kof + 16 * hh);
}

// C/D layout: lanes 0-15: (M=r, N=lane); lanes 16-31: (M=r+8, N=lane-16).

// ---------------- kernels ---------------------------------------------------

// Convert the five 256x256 fp32 weights to bf16 (consecutive in ws).
__global__ void k_prep_weights(const float* __restrict__ wp, const float* __restrict__ wq,
                               const float* __restrict__ wk, const float* __restrict__ wv,
                               const float* __restrict__ wo, __bf16* __restrict__ dst) {
  int i = blockIdx.x * blockDim.x + threadIdx.x;  // 0 .. 5*65536-1
  int which = i >> 16, r = i & 65535;
  const float* s = which == 0 ? wp : which == 1 ? wq : which == 2 ? wk
                                   : which == 3 ? wv : wo;
  dst[i] = (__bf16)s[r];
}

// Zero the pad token slots (l = 729..735) of h/q/k (row-major [l][d]),
// vT (transposed [d][l]) and ctx.
__global__ void k_zero_pads(__bf16* __restrict__ h, __bf16* __restrict__ q,
                            __bf16* __restrict__ k, __bf16* __restrict__ vT,
                            __bf16* __restrict__ c) {
  int i = blockIdx.x * blockDim.x + threadIdx.x;
  const int per = BATCH * 7 * D;  // 114688
  if (i >= 5 * per) return;
  int arr = i / per, r = i % per;
  if (arr == 3) {
    // vT[b][d][729..735]
    int b = r / (D * 7), rem = r % (D * 7);
    int d = rem / 7, l = LR + rem % 7;
    vT[((size_t)b * D + d) * LP + l] = (__bf16)0.f;
  } else {
    int b = r / (7 * D), rem = r % (7 * D);
    int l = LR + rem / D, d = rem % D;
    __bf16* dst = arr == 0 ? h : arr == 1 ? q : arr == 2 ? k : c;
    dst[((size_t)b * LP + l) * D + d] = (__bf16)0.f;
  }
}

// Fused unfold(16x16,stride 8) + patch projection: h = patches @ w_patch^T.
__global__ void k_patch_proj(const float* __restrict__ x, const __bf16* __restrict__ wp,
                             __bf16* __restrict__ h) {
  __shared__ __bf16 At[16 * 256];
  int qt = blockIdx.x, b = blockIdx.y;
  for (int i = threadIdx.x; i < 16 * 256; i += 256) {
    int m = i >> 8, f = i & 255;
    int l = qt * 16 + m;
    float v = 0.f;
    if (l < LR) {
      int py = l / GP, px = l % GP;
      int pr = f >> 4, pc = f & 15;
      v = x[(size_t)b * IMG * IMG + (size_t)(py * 8 + pr) * IMG + (px * 8 + pc)];
    }
    At[i] = (__bf16)v;
  }
  __syncthreads();
  int lane = LANE, w = WAVE;
  int n0a = w * 16, n0b = (w + 8) * 16;
  v8f acc0 = {}, acc1 = {};
#pragma unroll
  for (int kc = 0; kc < 8; ++kc) {
    v16bf a  = load_a_frag(At, 0, 256, kc * 32, lane);
    v16bf b0 = load_bT_frag(wp, n0a, 256, kc * 32, lane);
    v16bf b1 = load_bT_frag(wp, n0b, 256, kc * 32, lane);
    acc0 = wmma_bf16(a, b0, acc0);
    acc1 = wmma_bf16(a, b1, acc1);
  }
  int n = lane & 15, hh = lane >> 4;
#pragma unroll
  for (int r = 0; r < 8; ++r) {
    int m = r + 8 * hh, l = qt * 16 + m;
    if (l < LR) {
      size_t base = ((size_t)b * LP + l) * D;
      h[base + n0a + n] = (__bf16)acc0[r];
      h[base + n0b + n] = (__bf16)acc1[r];
    }
  }
}

// q/k = h @ w{q,k}^T (row-major [l][d]); v stored TRANSPOSED: vT[b][d][l].
// The 1/sqrt(dk)=1/16 softmax scale is folded into q.
__global__ void k_qkv(const __bf16* __restrict__ h, const __bf16* __restrict__ wq,
                      const __bf16* __restrict__ wk, const __bf16* __restrict__ wv,
                      __bf16* __restrict__ q, __bf16* __restrict__ k,
                      __bf16* __restrict__ vT) {
  __shared__ __bf16 Ht[16 * 256];
  int qt = blockIdx.x, b = blockIdx.y;
  const uint4* src = (const uint4*)(h + ((size_t)b * LP + qt * 16) * D);
  uint4* dst = (uint4*)Ht;
  dst[threadIdx.x]       = src[threadIdx.x];
  dst[threadIdx.x + 256] = src[threadIdx.x + 256];
  __syncthreads();
  int lane = LANE, w = WAVE;
  // hoist the shared A fragments once per wave, reuse for all 6 output tiles
  v16bf hfr[8];
#pragma unroll
  for (int kc = 0; kc < 8; ++kc) hfr[kc] = load_a_frag(Ht, 0, 256, kc * 32, lane);
#pragma unroll
  for (int t = 0; t < 6; ++t) {
    int tile = w * 6 + t;              // 48 tiles: [q:0-15][k:16-31][v:32-47]
    int mat = tile >> 4, dt = tile & 15;
    const __bf16* W = mat == 0 ? wq : mat == 1 ? wk : wv;
    float scl = mat == 0 ? 0.0625f : 1.0f;
    v8f acc = {};
#pragma unroll
    for (int kc = 0; kc < 8; ++kc) {
      v16bf bb = load_bT_frag(W, dt * 16, 256, kc * 32, lane);
      acc = wmma_bf16(hfr[kc], bb, acc);
    }
    int n = lane & 15, hh = lane >> 4;
    if (mat < 2) {
      __bf16* O = mat == 0 ? q : k;
#pragma unroll
      for (int r = 0; r < 8; ++r) {
        int m = r + 8 * hh, l = qt * 16 + m;
        if (l < LR) O[((size_t)b * LP + l) * D + dt * 16 + n] = (__bf16)(acc[r] * scl);
      }
    } else {
      // transposed store: lane writes 8 consecutive l values of channel dt*16+n
#pragma unroll
      for (int r = 0; r < 8; ++r) {
        int m = r + 8 * hh, l = qt * 16 + m;
        if (l < LR) vT[((size_t)b * D + dt * 16 + n) * LP + l] = (__bf16)acc[r];
      }
    }
  }
}

// Attention for one (batch, 16-query tile): S = qK^T (bf16 LDS, pad keys
// masked), in-place softmax, ctx = P V (V transposed -> contiguous B frags).
__global__ void k_attn(const __bf16* __restrict__ qm, const __bf16* __restrict__ km,
                       const __bf16* __restrict__ vT, __bf16* __restrict__ ctx) {
  __shared__ __bf16 Sb[16 * LP];   // scores, then probabilities in place
  __shared__ float sinv[16];
  int qt = blockIdx.x, b = blockIdx.y;
  int lane = LANE, w = WAVE;
  const __bf16* qbase  = qm + (size_t)b * LP * D;
  const __bf16* kbase  = km + (size_t)b * LP * D;
  const __bf16* vTbase = vT + (size_t)b * D * LP;

  // hoist the 8 q fragments (shared across all key tiles of this wave)
  v16bf qfr[8];
#pragma unroll
  for (int kc = 0; kc < 8; ++kc) qfr[kc] = load_a_frag(qbase, qt * 16, D, kc * 32, lane);

  // phase 1: scores, 8 waves strided over the 46 key tiles
  for (int jt = w; jt < NT; jt += 8) {
    if (jt + 8 < NT)  // prefetch next key tile (global_prefetch_b8)
      __builtin_prefetch((const void*)(kbase + (size_t)(jt + 8) * 16 * D), 0, 1);
    v8f acc = {};
#pragma unroll
    for (int kc = 0; kc < 8; ++kc) {
      v16bf bb = load_bT_frag(kbase, jt * 16, D, kc * 32, lane);
      acc = wmma_bf16(qfr[kc], bb, acc);
    }
    int n = lane & 15, hh = lane >> 4;
#pragma unroll
    for (int r = 0; r < 8; ++r) {
      int m = r + 8 * hh, col = jt * 16 + n;
      Sb[m * LP + col] = (col < LR) ? (__bf16)acc[r] : (__bf16)(-1e30f);
    }
  }
  __syncthreads();

  // phase 2: softmax, 2 rows per wave, wave-shuffle reductions
#pragma unroll
  for (int rr = 0; rr < 2; ++rr) {
    int m = w * 2 + rr;
    float mx = -1e30f;
    for (int c = lane; c < LP; c += 32) mx = fmaxf(mx, (float)Sb[m * LP + c]);
#pragma unroll
    for (int o = 16; o > 0; o >>= 1) mx = fmaxf(mx, __shfl_xor(mx, o, 32));
    float sum = 0.f;
    for (int c = lane; c < LP; c += 32) {
      float p = __expf((float)Sb[m * LP + c] - mx);
      Sb[m * LP + c] = (__bf16)p;   // in-place: same element, same address
      sum += p;
    }
#pragma unroll
    for (int o = 16; o > 0; o >>= 1) sum += __shfl_xor(sum, o, 32);
    if (lane == 0) sinv[m] = 1.0f / sum;
  }
  __syncthreads();

  // phase 3: ctx = P @ V, each wave owns two 16-wide dv tiles, K=736.
  // vT rows are dv channels -> B fragment is one contiguous 32B vector load.
  int n0a = w * 16, n0b = (w + 8) * 16;
  v8f acc0 = {}, acc1 = {};
  for (int kc = 0; kc < LP / 32; ++kc) {
    v16bf a  = load_a_frag(Sb, 0, LP, kc * 32, lane);
    v16bf b0 = load_bT_frag(vTbase, n0a, LP, kc * 32, lane);
    v16bf b1 = load_bT_frag(vTbase, n0b, LP, kc * 32, lane);
    acc0 = wmma_bf16(a, b0, acc0);
    acc1 = wmma_bf16(a, b1, acc1);
  }
  int n = lane & 15, hh = lane >> 4;
#pragma unroll
  for (int r = 0; r < 8; ++r) {
    int m = r + 8 * hh, l = qt * 16 + m;
    if (l < LR) {
      float s = sinv[m];
      size_t base = ((size_t)b * LP + l) * D;
      ctx[base + n0a + n] = (__bf16)(acc0[r] * s);
      ctx[base + n0b + n] = (__bf16)(acc1[r] * s);
    }
  }
}

// mean over tokens (mean commutes with the linear wo projection)
__global__ void k_ctxmean(const __bf16* __restrict__ ctx, float* __restrict__ cm) {
  int b = blockIdx.x, d = threadIdx.x;
  const __bf16* p = ctx + (size_t)b * LP * D + d;
  float s = 0.f;
  for (int l = 0; l < LR; ++l) s += (float)p[(size_t)l * D];
  cm[b * D + d] = s * (1.0f / 729.0f);
}

// pooled = ctxmean @ wo^T  (64x256 @ 256x256) — one workgroup, 64 WMMA tiles
__global__ void k_pool_proj(const float* __restrict__ cm, const __bf16* __restrict__ wo,
                            float* __restrict__ pooled) {
  int lane = LANE, w = WAVE;
#pragma unroll
  for (int t = 0; t < 8; ++t) {
    int tile = w * 8 + t;             // 0..63
    int mt = tile >> 4, dt = tile & 15;
    v8f acc = {};
#pragma unroll
    for (int kc = 0; kc < 8; ++kc) {
      v16bf a  = load_a_frag_f32(cm, mt * 16, D, kc * 32, lane);
      v16bf bb = load_bT_frag(wo, dt * 16, D, kc * 32, lane);
      acc = wmma_bf16(a, bb, acc);
    }
    int n = lane & 15, hh = lane >> 4;
#pragma unroll
    for (int r = 0; r < 8; ++r) {
      int m = mt * 16 + r + 8 * hh;
      pooled[m * D + dt * 16 + n] = acc[r];
    }
  }
}

// out = pooled @ w_final^T + b_final (64x10, keep fp32)
__global__ void k_final(const float* __restrict__ pooled, const float* __restrict__ wf,
                        const float* __restrict__ bfin, float* __restrict__ out) {
  int i = blockIdx.x * blockDim.x + threadIdx.x;
  if (i >= BATCH * 10) return;
  int b = i / 10, c = i % 10;
  float s = bfin[c];
  const float* p = pooled + b * D;
  const float* wr = wf + c * D;
  for (int d = 0; d < D; ++d) s += p[d] * wr[d];
  out[i] = s;
}

// ---------------- launch ----------------------------------------------------

extern "C" void kernel_launch(void* const* d_in, const int* in_sizes, int n_in,
                              void* d_out, int out_size, void* d_ws, size_t ws_size,
                              hipStream_t stream) {
  const float* x  = (const float*)d_in[0];
  const float* wp = (const float*)d_in[1];
  const float* wq = (const float*)d_in[2];
  const float* wk = (const float*)d_in[3];
  const float* wv = (const float*)d_in[4];
  const float* wo = (const float*)d_in[5];
  const float* wf = (const float*)d_in[6];
  const float* bf = (const float*)d_in[7];
  float* out = (float*)d_out;

  char* ws = (char*)d_ws;
  const size_t WSZ = 65536;                   // one 256x256 weight, elements
  const size_t TOK = (size_t)BATCH * LP * D;  // padded token tensor, elements
  __bf16* wbf    = (__bf16*)ws;               // 5 weights, consecutive
  __bf16* wp_bf  = wbf;
  __bf16* wq_bf  = wbf + 1 * WSZ;
  __bf16* wk_bf  = wbf + 2 * WSZ;
  __bf16* wv_bf  = wbf + 3 * WSZ;
  __bf16* wo_bf  = wbf + 4 * WSZ;
  size_t off = 5 * WSZ * sizeof(__bf16);
  __bf16* h_bf   = (__bf16*)(ws + off); off += TOK * sizeof(__bf16);
  __bf16* q_bf   = (__bf16*)(ws + off); off += TOK * sizeof(__bf16);
  __bf16* k_bf   = (__bf16*)(ws + off); off += TOK * sizeof(__bf16);
  __bf16* vT_bf  = (__bf16*)(ws + off); off += TOK * sizeof(__bf16);  // [b][d][l]
  __bf16* ctx_bf = (__bf16*)(ws + off); off += TOK * sizeof(__bf16);
  float*  cmean  = (float*)(ws + off);  off += (size_t)BATCH * D * sizeof(float);
  float*  pooled = (float*)(ws + off);  off += (size_t)BATCH * D * sizeof(float);
  (void)ws_size; (void)in_sizes; (void)n_in; (void)out_size;

  k_prep_weights<<<1280, 256, 0, stream>>>(wp, wq, wk, wv, wo, wbf);
  k_zero_pads<<<2240, 256, 0, stream>>>(h_bf, q_bf, k_bf, vT_bf, ctx_bf);

  dim3 g(NT, BATCH);
  k_patch_proj<<<g, 256, 0, stream>>>(x, wp_bf, h_bf);
  k_qkv<<<g, 256, 0, stream>>>(h_bf, wq_bf, wk_bf, wv_bf, q_bf, k_bf, vT_bf);
  k_attn<<<g, 256, 0, stream>>>(q_bf, k_bf, vT_bf, ctx_bf);
  k_ctxmean<<<BATCH, 256, 0, stream>>>(ctx_bf, cmean);
  k_pool_proj<<<1, 256, 0, stream>>>(cmean, wo_bf, pooled);
  k_final<<<3, 256, 0, stream>>>(pooled, wf, bf, out);
}
